// RNNTJoint_18476949307461
// MI455X (gfx1250) — compile-verified, matrix operand order checked by reading, not run on hardware
//
#include <hip/hip_runtime.h>
#include <hip/hip_bf16.h>

typedef _Float16 v16h __attribute__((ext_vector_type(16)));
typedef _Float16 v8h  __attribute__((ext_vector_type(8)));
typedef _Float16 h4   __attribute__((ext_vector_type(4)));
typedef float    v8f  __attribute__((ext_vector_type(8)));

#define B_  4
#define T_  256
#define U_  128
#define EH_ 1024
#define PH_ 640
#define JH_ 640
#define V_  1024

__device__ __forceinline__ v16h make16(v8h lo, v8h hi) {
    return __builtin_shufflevector(lo, hi, 0,1,2,3,4,5,6,7,8,9,10,11,12,13,14,15);
}

// ---------------------------------------------------------------------------
// Projection: OUT[b][m][j] (f16) = bias[j] + sum_k SRC[b][k][m] * W[j][k]
// SRC is channels-first [B][K][M]; W is [J][K] row-major. Tiled f32 GEMM.
// Grid: (J/64, M/16, B), block 256.
// ---------------------------------------------------------------------------
__global__ __launch_bounds__(256) void proj_f16_kernel(
    const float* __restrict__ src, const float* __restrict__ W,
    const float* __restrict__ bias, _Float16* __restrict__ out,
    int K, int M, int J)
{
    __shared__ float sW[64][65];
    __shared__ float sS[64][17];

    const int b   = blockIdx.z;
    const int m0  = blockIdx.y * 16;
    const int j0  = blockIdx.x * 64;
    const int tid = threadIdx.x;
    const int jj  = tid & 63;
    const int tq  = tid >> 6;   // 0..3

    float acc[4] = {0.f, 0.f, 0.f, 0.f};

    for (int k0 = 0; k0 < K; k0 += 64) {
        // stage W tile (64j x 64k)
        {
            int r = tid >> 2;
            int c = (tid & 3) * 16;
            const float4* p = (const float4*)(W + (size_t)(j0 + r) * K + k0 + c);
            #pragma unroll
            for (int v = 0; v < 4; ++v) {
                float4 x = p[v];
                sW[r][c + 4*v + 0] = x.x;
                sW[r][c + 4*v + 1] = x.y;
                sW[r][c + 4*v + 2] = x.z;
                sW[r][c + 4*v + 3] = x.w;
            }
        }
        // stage SRC tile (64k x 16m)
        {
            int kk = tid >> 2;
            int mm = (tid & 3) * 4;
            float4 s = *(const float4*)(src + ((size_t)b * K + k0 + kk) * M + m0 + mm);
            sS[kk][mm + 0] = s.x;
            sS[kk][mm + 1] = s.y;
            sS[kk][mm + 2] = s.z;
            sS[kk][mm + 3] = s.w;
        }
        __syncthreads();
        #pragma unroll 8
        for (int kk = 0; kk < 64; ++kk) {
            float w = sW[jj][kk];
            acc[0] = fmaf(w, sS[kk][tq + 0],  acc[0]);
            acc[1] = fmaf(w, sS[kk][tq + 4],  acc[1]);
            acc[2] = fmaf(w, sS[kk][tq + 8],  acc[2]);
            acc[3] = fmaf(w, sS[kk][tq + 12], acc[3]);
        }
        __syncthreads();
    }

    float bb = bias[j0 + jj];
    #pragma unroll
    for (int i = 0; i < 4; ++i) {
        int m = m0 + tq + 4 * i;
        out[((size_t)b * M + m) * J + j0 + jj] = (_Float16)(acc[i] + bb);
    }
}

// f32 -> f16 cast (vector of 4)
__global__ void cvt_f16_kernel(const float* __restrict__ in,
                               _Float16* __restrict__ outp, int n4)
{
    int i = blockIdx.x * blockDim.x + threadIdx.x;
    if (i < n4) {
        float4 x = ((const float4*)in)[i];
        h4 h;
        h[0] = (_Float16)x.x; h[1] = (_Float16)x.y;
        h[2] = (_Float16)x.z; h[3] = (_Float16)x.w;
        ((h4*)outp)[i] = h;
    }
}

// ---------------------------------------------------------------------------
// Fused joint: for one (b, t, 32-u tile), build h = relu(E[t]+P[u]) in LDS,
// GEMM against W_out^T (V=1024) with v_wmma_f32_16x16x32_f16, fused
// log_softmax over V, nontemporal f32 stores.
// Block = 256 threads (8 waves). Wave w owns V columns [w*128, w*128+128).
// Accumulators: 2 M-tiles x 8 N-tiles of v8f (128 VGPRs).
// K-loop manually unrolled x2: B fragments pipelined one tile ahead, and the
// first half-step issues WGP-scope global_prefetch_b8 (inline asm, immediate
// offsets) so each 128B line of W_out is pulled into WGP$ two lines ahead.
// Grid: (U/32, T, B).
// ---------------------------------------------------------------------------
__global__ __launch_bounds__(256) void joint_kernel(
    const _Float16* __restrict__ Eh,   // [B][T][JH]
    const _Float16* __restrict__ Ph,   // [B][U][JH]
    const _Float16* __restrict__ Wh,   // [V][JH]
    const float*    __restrict__ bout, // [V]
    float*          __restrict__ out)  // [B][T][U][V]
{
    constexpr int HP = JH_ + 8;                      // padded f16 row pitch
    __shared__ __align__(16) _Float16 hbuf[32 * HP]; // 41.5 KB
    __shared__ float red_max[32][8];
    __shared__ float red_sum[32][8];

    const int u0 = blockIdx.x * 32;
    const int t  = blockIdx.y;
    const int b  = blockIdx.z;

    const int tid  = threadIdx.x;
    const int lane = tid & 31;
    const int wave = tid >> 5;     // 0..7
    const int l16  = lane & 15;
    const int half = lane >> 4;    // 0 or 1
    const int ksel = half * 8;     // A/B operand K-sub-select per lane half

    // ---- stage h = relu(E[b,t,:] + P[b,u0+r,:]) as f16 into LDS ----
    const _Float16* Erow = Eh + ((size_t)b * T_ + t) * JH_;
    const _Float16* Prow = Ph + ((size_t)b * U_ + u0) * JH_;
    #pragma unroll
    for (int it = 0; it < 10; ++it) {
        int chunk = tid + it * 256;       // 0..2559 -> 32 rows x 80 chunks of 8
        int r  = chunk / 80;
        int kc = (chunk % 80) * 8;
        v8h ev = *(const v8h*)(Erow + kc);
        v8h pv = *(const v8h*)(Prow + (size_t)r * JH_ + kc);
        v8h hv;
        #pragma unroll
        for (int i = 0; i < 8; ++i) {
            float x = (float)ev[i] + (float)pv[i];
            hv[i] = (_Float16)(x > 0.f ? x : 0.f);
        }
        *(v8h*)&hbuf[r * HP + kc] = hv;
    }
    __syncthreads();

    // ---- main GEMM over K = JH ----
    v8f c[2][8];
    #pragma unroll
    for (int mi = 0; mi < 2; ++mi)
        #pragma unroll
        for (int ni = 0; ni < 8; ++ni)
            #pragma unroll
            for (int d = 0; d < 8; ++d) c[mi][ni][d] = 0.f;

    const _Float16* wbase = Wh + ((size_t)(wave * 128 + l16)) * JH_ + ksel;
    const _Float16* arow0 = &hbuf[(0 * 16 + l16) * HP + ksel];
    const _Float16* arow1 = &hbuf[(1 * 16 + l16) * HP + ksel];

    for (int k0 = 0; k0 < JH_; k0 += 64) {
        // ======== half-step A: k = k0 (with WGP prefetch, 256B ahead) ========
        {
            const int kk = k0;
            v16h a0 = make16(*(const v8h*)(arow0 + kk), *(const v8h*)(arow0 + kk + 16));
            v16h a1 = make16(*(const v8h*)(arow1 + kk), *(const v8h*)(arow1 + kk + 16));
            const _Float16* wp0 = wbase + kk;
            v16h bcur = make16(*(const v8h*)wp0, *(const v8h*)(wp0 + 16));
            const _Float16* pfb = wbase + kk + 128;   // one shared base VGPR pair
            #pragma unroll
            for (int ni = 0; ni < 8; ++ni) {
                v16h bnxt = bcur;
                if (ni < 7) {
                    const _Float16* wp = wbase + (size_t)(ni + 1) * 16 * JH_ + kk;
                    bnxt = make16(*(const v8h*)wp, *(const v8h*)(wp + 16));
                }
                // scope omitted -> SCOPE 0 (WGP): fills all cache levels
                asm volatile("global_prefetch_b8 %0, off offset:%1"
                             :: "v"(pfb), "i"(ni * 16 * JH_ * 2));
                c[0][ni] = __builtin_amdgcn_wmma_f32_16x16x32_f16(
                    false, a0, false, bcur, (short)0, c[0][ni], false, false);
                c[1][ni] = __builtin_amdgcn_wmma_f32_16x16x32_f16(
                    false, a1, false, bcur, (short)0, c[1][ni], false, false);
                bcur = bnxt;
            }
        }
        // ======== half-step B: k = k0 + 32 (no prefetch) ========
        {
            const int kk = k0 + 32;
            v16h a0 = make16(*(const v8h*)(arow0 + kk), *(const v8h*)(arow0 + kk + 16));
            v16h a1 = make16(*(const v8h*)(arow1 + kk), *(const v8h*)(arow1 + kk + 16));
            const _Float16* wp0 = wbase + kk;
            v16h bcur = make16(*(const v8h*)wp0, *(const v8h*)(wp0 + 16));
            #pragma unroll
            for (int ni = 0; ni < 8; ++ni) {
                v16h bnxt = bcur;
                if (ni < 7) {
                    const _Float16* wp = wbase + (size_t)(ni + 1) * 16 * JH_ + kk;
                    bnxt = make16(*(const v8h*)wp, *(const v8h*)(wp + 16));
                }
                c[0][ni] = __builtin_amdgcn_wmma_f32_16x16x32_f16(
                    false, a0, false, bcur, (short)0, c[0][ni], false, false);
                c[1][ni] = __builtin_amdgcn_wmma_f32_16x16x32_f16(
                    false, a1, false, bcur, (short)0, c[1][ni], false, false);
                bcur = bnxt;
            }
        }
    }

    // ---- epilogue: + b_out, fused log_softmax over V ----
    const int vbase = wave * 128 + l16;
    float bo[8];
    #pragma unroll
    for (int ni = 0; ni < 8; ++ni) bo[ni] = bout[vbase + ni * 16];
    #pragma unroll
    for (int mi = 0; mi < 2; ++mi)
        #pragma unroll
        for (int ni = 0; ni < 8; ++ni)
            #pragma unroll
            for (int d = 0; d < 8; ++d) c[mi][ni][d] += bo[ni];

    // C layout: element d of lane -> row m = mi*16 + d + 8*half, col = vbase + ni*16
    float rmax[2][8];
    #pragma unroll
    for (int mi = 0; mi < 2; ++mi) {
        #pragma unroll
        for (int d = 0; d < 8; ++d) {
            float m = c[mi][0][d];
            #pragma unroll
            for (int ni = 1; ni < 8; ++ni) m = fmaxf(m, c[mi][ni][d]);
            #pragma unroll
            for (int off = 8; off >= 1; off >>= 1)
                m = fmaxf(m, __shfl_xor(m, off, 32));   // stays within 16-lane half
            int row = mi * 16 + d + 8 * half;
            if (l16 == 0) red_max[row][wave] = m;
        }
    }
    __syncthreads();
    #pragma unroll
    for (int mi = 0; mi < 2; ++mi)
        #pragma unroll
        for (int d = 0; d < 8; ++d) {
            int row = mi * 16 + d + 8 * half;
            float m = red_max[row][0];
            #pragma unroll
            for (int w = 1; w < 8; ++w) m = fmaxf(m, red_max[row][w]);
            rmax[mi][d] = m;
        }

    #pragma unroll
    for (int mi = 0; mi < 2; ++mi) {
        #pragma unroll
        for (int d = 0; d < 8; ++d) {
            float s = 0.f;
            #pragma unroll
            for (int ni = 0; ni < 8; ++ni) s += __expf(c[mi][ni][d] - rmax[mi][d]);
            #pragma unroll
            for (int off = 8; off >= 1; off >>= 1)
                s += __shfl_xor(s, off, 32);
            int row = mi * 16 + d + 8 * half;
            if (l16 == 0) red_sum[row][wave] = s;
        }
    }
    __syncthreads();

    const size_t obase = ((size_t)b * T_ + t) * U_ * (size_t)V_;
    #pragma unroll
    for (int mi = 0; mi < 2; ++mi) {
        #pragma unroll
        for (int d = 0; d < 8; ++d) {
            int row = mi * 16 + d + 8 * half;
            float s = red_sum[row][0];
            #pragma unroll
            for (int w = 1; w < 8; ++w) s += red_sum[row][w];
            float logZ = rmax[mi][d] + __logf(s);
            size_t rb = obase + (size_t)(u0 + row) * V_ + vbase;
            #pragma unroll
            for (int ni = 0; ni < 8; ++ni)
                __builtin_nontemporal_store(c[mi][ni][d] - logZ, out + rb + ni * 16);
        }
    }
}

// ---------------------------------------------------------------------------
extern "C" void kernel_launch(void* const* d_in, const int* in_sizes, int n_in,
                              void* d_out, int out_size, void* d_ws, size_t ws_size,
                              hipStream_t stream)
{
    (void)in_sizes; (void)n_in; (void)out_size; (void)ws_size;
    const float* enc    = (const float*)d_in[0];
    const float* dec    = (const float*)d_in[1];
    const float* W_enc  = (const float*)d_in[2];
    const float* b_enc  = (const float*)d_in[3];
    const float* W_pred = (const float*)d_in[4];
    const float* b_pred = (const float*)d_in[5];
    const float* W_out  = (const float*)d_in[6];
    const float* b_out  = (const float*)d_in[7];
    float* out = (float*)d_out;

    char* ws = (char*)d_ws;
    const size_t szE = (size_t)B_ * T_ * JH_ * sizeof(_Float16); // 1.31 MB
    const size_t szP = (size_t)B_ * U_ * JH_ * sizeof(_Float16); // 0.66 MB
    _Float16* Eh = (_Float16*)ws;
    _Float16* Ph = (_Float16*)(ws + szE);
    _Float16* Wh = (_Float16*)(ws + szE + szP);

    dim3 blk(256);
    // E[b,t,j] = b_enc[j] + sum_h enc[b,h,t] * W_enc[j,h]
    proj_f16_kernel<<<dim3(JH_ / 64, T_ / 16, B_), blk, 0, stream>>>(
        enc, W_enc, b_enc, Eh, EH_, T_, JH_);
    // P[b,u,j] = b_pred[j] + sum_h dec[b,h,u] * W_pred[j,h]
    proj_f16_kernel<<<dim3(JH_ / 64, U_ / 16, B_), blk, 0, stream>>>(
        dec, W_pred, b_pred, Ph, PH_, U_, JH_);
    // W_out -> f16
    int n4 = (V_ * JH_) / 4;
    cvt_f16_kernel<<<dim3((n4 + 255) / 256), blk, 0, stream>>>(W_out, Wh, n4);
    // fused joint + log_softmax
    joint_kernel<<<dim3(U_ / 32, T_, B_), blk, 0, stream>>>(Eh, Ph, Wh, b_out, out);
}
